// AMRNL_86466281603786
// MI455X (gfx1250) — matrix operand order repaired and checked by reading, not compiled, not run on hardware
//
#include <hip/hip_runtime.h>
#include <hip/hip_bf16.h>

typedef __attribute__((ext_vector_type(16))) _Float16 v16h;
typedef __attribute__((ext_vector_type(8)))  _Float16 v8h;
typedef __attribute__((ext_vector_type(4)))  _Float16 v4h;
typedef __attribute__((ext_vector_type(8)))  float    v8f;

// global-address-space view of a 32B weight fragment -> global_load_b128 x2
typedef __attribute__((address_space(1))) const v16h g1v16h;

#define B_  256
#define L_  100
#define E_  300
#define EP_ 320      // E padded to multiple of 32 (WMMA K)
#define H_  256
#define U_  10000
#define K_  32
#define G4_ 1024     // 4*H

__device__ __forceinline__ float sigf(float x) { return 1.0f / (1.0f + expf(-x)); }

// ---------------------------------------------------------------------------
// Kernel 1: convert LSTM weights to f16, zero-padding E 300 -> 320.
// Row-major per gate-column j, so a WMMA B-fragment slice (column j,
// K = e0+kh*16 .. +15) is one contiguous 32B load per lane.
// ---------------------------------------------------------------------------
__global__ void prep_weights_kernel(const float* __restrict__ Wih,
                                    const float* __restrict__ Whh,
                                    _Float16* __restrict__ Wih16,
                                    _Float16* __restrict__ Whh16)
{
    const int n1 = G4_ * EP_;
    const int n2 = G4_ * H_;
    for (int i = blockIdx.x * blockDim.x + threadIdx.x; i < n1 + n2;
         i += gridDim.x * blockDim.x) {
        if (i < n1) {
            int j = i / EP_, e = i % EP_;
            Wih16[i] = (e < E_) ? (_Float16)Wih[j * E_ + e] : (_Float16)0.0f;
        } else {
            int i2 = i - n1;
            Whh16[i2] = (_Float16)Whh[i2];
        }
    }
}

// ---------------------------------------------------------------------------
// Kernel 2: fused gather + 100-step LSTM, one 16-row batch tile per WG.
// grid = 32 (16 tiles x {question, answer}), block = 256 (8 waves).
// - Each wave owns h-indices [w*32,w*32+32) in all 4 gate quarters, so the
//   cell update is register-local; c stays in VGPRs, h round-trips LDS as f16.
// - Weight fragments: offset laundered via empty volatile asm (defeats LICM /
//   the round-1 scratch spill) and dereferenced through addrspace(1) so they
//   lower to global_load_b128 (LOADcnt only), decoupled from LDS DScnt waits.
// - x-gather software-pipelined: step t+1's word2vec rows are prefetched into
//   registers while step t's 144 WMMAs run, stored to LDS next iteration.
// ---------------------------------------------------------------------------
__global__ void __launch_bounds__(256)
lstm_kernel(const int*  __restrict__ qlist,  const int* __restrict__ alist,
            const int*  __restrict__ ctable, const float* __restrict__ w2v,
            const _Float16* __restrict__ Wih16,
            const _Float16* __restrict__ Whh16,
            const float* __restrict__ bias,
            float* __restrict__ qh, float* __restrict__ ah)
{
    __shared__ __align__(32) _Float16 xbuf[16][EP_];  // gathered x_t (f16)
    __shared__ __align__(32) _Float16 hbuf[16][H_];   // h_{t-1} (f16)
    __shared__ int cids[16];

    const int tid  = threadIdx.x;
    const int lane = tid & 31;
    const int wv   = tid >> 5;      // wave id 0..7
    const int kh   = lane >> 4;     // K-half selector (A frag / B frag / C rows)
    const int nrel = lane & 15;     // column within 16-wide N tile
    const int mrow = lane & 15;     // A-fragment source row

    const int which = blockIdx.x >> 4;        // 0 = question, 1 = answer
    const int m0    = (blockIdx.x & 15) * 16; // batch-row base
    const int* idlist = which ? alist : qlist;
    float*     hout   = which ? ah : qh;

    if (tid < 16) cids[tid] = idlist[m0 + tid] - U_;
    for (int i = tid; i < 16 * H_; i += 256)
        ((_Float16*)hbuf)[i] = (_Float16)0.0f;

    const int grow  = tid >> 4;        // gather: row handled by this thread
    const int ebase = (tid & 15) * 20; // gather: 20 elems per thread per row
    const bool gact = (ebase < E_);    // the ebase==300 thread only owns zeros

    // static zero tail (E..EP): written once, never touched again
    if (!gact) {
        v4h z = {};
#pragma unroll
        for (int v = 0; v < 5; v++)
            *(v4h*)&xbuf[grow][ebase + v * 4] = z;
    }

    // Per-thread bias for accumulator init (gate q, subtile s, column nrel).
    float bias_reg[4][2];
#pragma unroll
    for (int q = 0; q < 4; q++)
#pragma unroll
        for (int s = 0; s < 2; s++)
            bias_reg[q][s] = bias[q * 256 + wv * 32 + s * 16 + nrel];

    float c_reg[2][8];
#pragma unroll
    for (int s = 0; s < 2; s++)
#pragma unroll
        for (int r = 0; r < 8; r++) c_reg[s][r] = 0.0f;

    __syncthreads();                       // cids / zero-tail visible
    const long myrowbase = (long)cids[grow] * L_;  // content row (reg-cached)

    // prologue: prefetch x_0
    float4 pf[5];
    if (gact) {
        int tok = ctable[myrowbase + 0];
        const float* wr = w2v + (long)tok * E_;
#pragma unroll
        for (int v = 0; v < 5; v++)
            pf[v] = *(const float4*)(wr + ebase + v * 4);
    }

    for (int t = 0; t < L_; t++) {
        // Launder an offset: the 144 weight-fragment loads below must stay
        // inside the loop (L2-resident stream), not hoist-and-spill.
        long lz = 0;
        asm volatile("" : "+s"(lz));
        const _Float16* wih_t = Wih16 + lz;
        const _Float16* whh_t = Whh16 + lz;

        __syncthreads();  // (a) prev matmul done -> safe to overwrite xbuf
        if (gact) {       // commit prefetched x_t
#pragma unroll
            for (int v = 0; v < 5; v++) {
                v4h hv;
                hv[0] = (_Float16)pf[v].x; hv[1] = (_Float16)pf[v].y;
                hv[2] = (_Float16)pf[v].z; hv[3] = (_Float16)pf[v].w;
                *(v4h*)&xbuf[grow][ebase + v * 4] = hv;
            }
        }
        __syncthreads();  // (b) xbuf ready; hbuf from previous update ready

        // prefetch x_{t+1}: loads fly while the WMMAs below execute
        if (gact && (t + 1 < L_)) {
            int tok = ctable[myrowbase + t + 1];
            const float* wr = w2v + (long)tok * E_;
#pragma unroll
            for (int v = 0; v < 5; v++)
                pf[v] = *(const float4*)(wr + ebase + v * 4);
        }

        // ---- gates = bias + x_t*Wih^T + h*Whh^T ----
        v8f acc[4][2];
#pragma unroll
        for (int q = 0; q < 4; q++)
#pragma unroll
            for (int s = 0; s < 2; s++) {
                v8f a;
#pragma unroll
                for (int r = 0; r < 8; r++) a[r] = bias_reg[q][s];
                acc[q][s] = a;
            }

        // x part: K = 320 -> 10 WMMA K-steps
#pragma unroll
        for (int kc = 0; kc < 10; kc++) {
            const int e0 = kc * 32;
            v16h af = *(const v16h*)&xbuf[mrow][e0 + kh * 16];
#pragma unroll
            for (int q = 0; q < 4; q++)
#pragma unroll
                for (int s = 0; s < 2; s++) {
                    const int col = q * 256 + wv * 32 + s * 16 + nrel;
                    v16h bf = *(g1v16h*)&wih_t[(long)col * EP_ + e0 + kh * 16];
                    acc[q][s] = __builtin_amdgcn_wmma_f32_16x16x32_f16(
                        false, af, false, bf, (short)0, acc[q][s], false, false);
                }
        }
        // h part: K = 256 -> 8 WMMA K-steps
#pragma unroll
        for (int kc = 0; kc < 8; kc++) {
            const int k0 = kc * 32;
            v16h af = *(const v16h*)&hbuf[mrow][k0 + kh * 16];
#pragma unroll
            for (int q = 0; q < 4; q++)
#pragma unroll
                for (int s = 0; s < 2; s++) {
                    const int col = q * 256 + wv * 32 + s * 16 + nrel;
                    v16h bf = *(g1v16h*)&whh_t[(long)col * H_ + k0 + kh * 16];
                    acc[q][s] = __builtin_amdgcn_wmma_f32_16x16x32_f16(
                        false, af, false, bf, (short)0, acc[q][s], false, false);
                }
        }
        __syncthreads();  // (c) all hbuf reads done -> safe to write new h

        // ---- cell update (register-local: all 4 gates in this wave) ----
#pragma unroll
        for (int s = 0; s < 2; s++) {
            const int j = wv * 32 + s * 16 + nrel;
#pragma unroll
            for (int r = 0; r < 8; r++) {
                const int m = r + 8 * kh;  // C/D layout: lanes>=16 hold M=8..15
                float ig = sigf(acc[0][s][r]);
                float fg = sigf(acc[1][s][r]);
                float gg = tanhf(acc[2][s][r]);
                float og = sigf(acc[3][s][r]);
                float c  = fg * c_reg[s][r] + ig * gg;
                c_reg[s][r] = c;
                float h = og * tanhf(c);
                hbuf[m][j] = (_Float16)h;
                if (t == L_ - 1) hout[(long)(m0 + m) * H_ + j] = h;
            }
        }
    }
}

// ---------------------------------------------------------------------------
// Kernel 3: bilinear match * cosine, and degree-normalized neighbor regular.
// grid = B (one block per batch row), block = 256 (one thread per h-dim).
// ---------------------------------------------------------------------------
__device__ __forceinline__ float block_reduce(float v, float* red)
{
    const int j = threadIdx.x;
    __syncthreads();           // protect previous red[0] consumers
    red[j] = v;
    __syncthreads();
    for (int s = 128; s > 0; s >>= 1) {
        if (j < s) red[j] += red[j + s];
        __syncthreads();
    }
    return red[0];
}

__global__ void __launch_bounds__(256)
final_kernel(const float* __restrict__ qh, const float* __restrict__ ah,
             const int*  __restrict__ user_list,
             const float* __restrict__ uet,     // user_embed_table [U+2, H]
             const int*  __restrict__ adj,      // [U+2, K]
             const float* __restrict__ degree,  // [U+2, 1]
             const float* __restrict__ bilW,    // [1, H, H]
             const float* __restrict__ bilb,    // [1]
             float* __restrict__ out)           // [B] result ++ [B*H] regular
{
    __shared__ float qs[256];
    __shared__ float red[256];
    const int b = blockIdx.x;
    const int j = threadIdx.x;
    const int uid = user_list[b];

    const float qv = qh[b * H_ + j];
    const float av = ah[b * H_ + j];
    const float uv = uet[(long)uid * H_ + j];
    qs[j] = qv;
    __syncthreads();

    // t[j] = sum_h qh[h] * W[h][j]   (then dot with a_h via reduction)
    float tg = 0.0f;
#pragma unroll 4
    for (int h = 0; h < H_; h++) tg += qs[h] * bilW[h * H_ + j];

    float match = block_reduce(tg * av, red) + bilb[0];
    float qu    = block_reduce(qv * uv, red);
    float qq    = block_reduce(qv * qv, red);
    float uu    = block_reduce(uv * uv, red);

    // neighbor aggregation
    float ns = 0.0f;
#pragma unroll
    for (int k = 0; k < K_; k++)
        ns += uet[(long)adj[uid * K_ + k] * H_ + j];
    float neigh = ns / (degree[uid] + 1.0f);   // SMOOTH = 1.0
    float diff  = uv - neigh;
    float dd    = block_reduce(diff * diff, red);
    float dn    = fmaxf(sqrtf(dd), 1e-12f);
    out[B_ + b * H_ + j] = diff / dn;

    if (j == 0) {
        float cosv = qu / (fmaxf(sqrtf(qq), 1e-8f) * fmaxf(sqrtf(uu), 1e-8f));
        out[b] = match * cosv;
    }
}

// ---------------------------------------------------------------------------
extern "C" void kernel_launch(void* const* d_in, const int* in_sizes, int n_in,
                              void* d_out, int out_size, void* d_ws, size_t ws_size,
                              hipStream_t stream)
{
    (void)in_sizes; (void)n_in; (void)out_size; (void)ws_size;

    const int*   question_list = (const int*)  d_in[0];
    const int*   answer_list   = (const int*)  d_in[1];
    const int*   user_list     = (const int*)  d_in[2];
    const float* word2vec      = (const float*)d_in[3];
    const int*   content_table = (const int*)  d_in[4];
    const float* user_embed    = (const float*)d_in[5];
    const int*   adj           = (const int*)  d_in[6];
    const float* degree        = (const float*)d_in[7];
    const float* Wih           = (const float*)d_in[8];
    const float* Whh           = (const float*)d_in[9];
    const float* bvec          = (const float*)d_in[10];
    const float* bilW          = (const float*)d_in[11];
    const float* bilb          = (const float*)d_in[12];

    // workspace layout
    char* ws = (char*)d_ws;
    _Float16* Wih16 = (_Float16*)(ws);                    // 1024*320*2 = 655360
    _Float16* Whh16 = (_Float16*)(ws + 655360);           // 1024*256*2 = 524288
    float*    qh    = (float*)   (ws + 1179648);          // 256*256*4  = 262144
    float*    ah    = (float*)   (ws + 1441792);          // 256*256*4  = 262144

    prep_weights_kernel<<<1024, 256, 0, stream>>>(Wih, Whh, Wih16, Whh16);
    lstm_kernel<<<32, 256, 0, stream>>>(question_list, answer_list,
                                        content_table, word2vec,
                                        Wih16, Whh16, bvec, qh, ah);
    final_kernel<<<B_, 256, 0, stream>>>(qh, ah, user_list, user_embed,
                                         adj, degree, bilW, bilb,
                                         (float*)d_out);
}
